// TransformerBlock_EA_13417477833318
// MI455X (gfx1250) — matrix operand order repaired and checked by reading, not compiled, not run on hardware
//
#include <hip/hip_runtime.h>
#include <hip/hip_bf16.h>

// ---------------------------------------------------------------------------
// CDNA5 (gfx1250) bf16 WMMA implementation of the EA-transformer + UnetResBlock.
// All GEMMs / convs run on v_wmma_f32_16x16x32_bf16 (f32 accumulate).
// Wave32; one 16-row output tile per wave.  k_kstats additionally exercises the
// Tensor Data Mover (tensor_load_to_lds + s_wait_tensorcnt) with LDS double
// buffering overlapped against an online-softmax reduction.
// ---------------------------------------------------------------------------

typedef __attribute__((ext_vector_type(16))) __bf16 v16bf;
typedef __attribute__((ext_vector_type(8)))  float  v8f;
typedef unsigned __attribute__((ext_vector_type(4))) v4u;
typedef int      __attribute__((ext_vector_type(4))) v4i;
typedef int      __attribute__((ext_vector_type(8))) v8i;

#define BATCH 2
#define CCH   128
#define NTOK  32768           // 32*32*32 per batch
#define MTOT  (BATCH * NTOK)  // 65536 tokens total
#define KLDA  136             // padded LDS row stride (bf16 elems) to avoid bank conflicts

// ---- TDM availability probes (device pass only; host pass falls back) ------
#if defined(__AMDGCN__) && defined(__has_builtin)
#  if __has_builtin(__builtin_amdgcn_tensor_load_to_lds) && \
      __has_builtin(__builtin_amdgcn_s_wait_tensorcnt)
#    define HAVE_TDM 1
#  else
#    define HAVE_TDM 0
#  endif
#else
#  define HAVE_TDM 0
#endif

#if HAVE_TDM && __has_include(<hip/amd_detail/amd_gfx1250_TDM.h>)
#  define TDM_SIX_ARG 1
#else
#  define TDM_SIX_ARG 0
#endif

union UF { v16bf v; unsigned u[8]; };

static __device__ inline unsigned ld2bf(const __bf16* p) { return *(const unsigned*)p; }

static __device__ inline unsigned pack2(float a, float b) {
  union { __bf16 h[2]; unsigned u; } t;
  t.h[0] = (__bf16)a; t.h[1] = (__bf16)b;
  return t.u;
}

static __device__ inline float lrelu(float v) { return v >= 0.f ? v : 0.01f * v; }

static __device__ inline v8f wmma_bf16(v16bf a, v16bf b, v8f c) {
  // D = A(16x32) * B(32x16) + C, f32 accumulate
  return __builtin_amdgcn_wmma_f32_16x16x32_bf16(false, a, false, b, (short)0, c,
                                                 false, false);
}

// A-matrix fragment (16x32 bf16) from LDS tile, row-major with stride lda.
// ISA layout: lane L -> row m = L&15; kb = (L>>4)*8;
//   VGPR j<4 : k = kb+2j,+1 ; VGPR j>=4 : k = 16+kb+2(j-4),+1
static __device__ inline v16bf frag_a_lds(const __bf16* A, int lda, int k0, int lane) {
  int m = lane & 15, kb = (lane >> 4) * 8;
  const __bf16* r = A + m * lda + k0 + kb;
  UF f;
#pragma unroll
  for (int j = 0; j < 4; ++j) {
    f.u[j]     = ld2bf(r + 2 * j);
    f.u[j + 4] = ld2bf(r + 16 + 2 * j);
  }
  return f.v;
}

// B-matrix fragment (32x16 bf16) from a row-major weight matrix W[o][c]
// (B[k][n] = W[n0+n][k0+k]).  Lane L -> col n = L&15; kh = (L>>4)*8;
//   VGPR j<4 : k = kh+2j,+1 ; VGPR j>=4 : k = 16+kh+2(j-4),+1
static __device__ inline v16bf frag_b_w(const __bf16* W, int ldb, int k0, int n0, int lane) {
  int n = lane & 15, kh = (lane >> 4) * 8;
  const __bf16* r = W + (size_t)(n0 + n) * ldb + k0 + kh;
  UF f;
#pragma unroll
  for (int j = 0; j < 8; ++j) f.u[j] = ld2bf(r + 2 * j + (j >= 4 ? 8 : 0));
  return f.v;
}

// ---------------------------------------------------------------------------
// Tensor Data Mover: 1-D DMA of n f32 elements from global into LDS.
// D# per CDNA5 ISA:
//   group0: [1:0] count=1 | [63:32] lds byte addr | [120:64] global addr | type=2
//   group1: data_size=4B; tensor_dim0 = tile_dim0 = n; tensor_dim1 = 1;
//           tensor_dim0_stride = n; no padding / multicast / iterate.
//   groups 2,3: zero (<=2D tensor)
// ---------------------------------------------------------------------------
#if HAVE_TDM
static __device__ inline unsigned lds_byte_off(const void* p) {
  // Flat LDS addresses carry the LDS byte offset in the low 32 bits.
  return (unsigned)(unsigned long long)(uintptr_t)p;
}

static __device__ inline void tdm_load_1d(unsigned lds_byte, const void* gsrc, unsigned n) {
  unsigned long long ga = (unsigned long long)(uintptr_t)gsrc;
  v4u g0 = { 1u,                                    // count=1, user descriptor
             lds_byte,                               // LDS destination (bytes)
             (unsigned)ga,                           // global addr [31:0]
             (unsigned)((ga >> 32) & 0x01ffffffu) | (2u << 30) };  // [56:32] | type=2
  v8i g1 = { (int)(2u << 16),                        // data_size=2 -> 4 bytes
             (int)((n & 0xffffu) << 16),             // tensor_dim0[15:0]  @ bits 63:48
             (int)(((n >> 16) & 0xffffu) | (1u << 16)), // dim0[31:16] | tensor_dim1=1
             (int)((n & 0xffffu) << 16),             // tile_dim0 @ bits 127:112
             0,                                      // tile_dim1 / tile_dim2 unused
             (int)n,                                 // tensor_dim0_stride[31:0]
             0, 0 };
  v4i gz = { 0, 0, 0, 0 };
#if TDM_SIX_ARG
  v8i gz8 = { 0, 0, 0, 0, 0, 0, 0, 0 };
  __builtin_amdgcn_tensor_load_to_lds(g0, g1, gz, gz, gz8, 0);
#else
  __builtin_amdgcn_tensor_load_to_lds(g0, g1, gz, gz, 0);
#endif
}
#endif  // HAVE_TDM

// ---------------------------------------------------------------------------
// Weight prep: f32 -> bf16; conv weights re-laid out [tap][oc][ic]
// ---------------------------------------------------------------------------
__global__ void k_prep(const float* __restrict__ wq, const float* __restrict__ wk,
                       const float* __restrict__ wv, const float* __restrict__ wrep,
                       const float* __restrict__ w8, const float* __restrict__ c1,
                       const float* __restrict__ c2,
                       __bf16* wqb, __bf16* wkb, __bf16* wvb, __bf16* wrepb, __bf16* w8b,
                       __bf16* c1b, __bf16* c2b) {
  int i = blockIdx.x * 256 + threadIdx.x;
  if (i < CCH * CCH) {
    wqb[i]   = (__bf16)wq[i];
    wkb[i]   = (__bf16)wk[i];
    wvb[i]   = (__bf16)wv[i];
    wrepb[i] = (__bf16)wrep[i];
    w8b[i]   = (__bf16)w8[i];
  }
  if (i < 27 * CCH * CCH) {
    int ic = i & 127, oc = (i >> 7) & 127, tap = i >> 14;
    size_t src = (size_t)(oc * CCH + ic) * 27 + tap;
    c1b[i] = (__bf16)c1[src];
    c2b[i] = (__bf16)c2[src];
  }
}

// ---------------------------------------------------------------------------
// LayerNorm + QKV GEMMs.  One wave per 16-token tile.  Q/K/V stored
// channel-major: X[(b*128+c)*32768 + n]
// ---------------------------------------------------------------------------
__global__ void k_ln_qkv(const float* __restrict__ x, const float* __restrict__ ln_g,
                         const float* __restrict__ ln_b,
                         const __bf16* __restrict__ wqb, const __bf16* __restrict__ wkb,
                         const __bf16* __restrict__ wvb,
                         float* __restrict__ Qc, float* __restrict__ Kc,
                         float* __restrict__ Vc) {
  __shared__ __bf16 At[8][16 * KLDA];
  int lane = threadIdx.x & 31, wave = threadIdx.x >> 5;
  int tile = blockIdx.x * 8 + wave;
  int g0 = tile * 16;
  int b = g0 >> 15, n0 = g0 & (NTOK - 1);
  int m = lane & 15, half = lane >> 4;

  const float* xb = x + (size_t)b * CCH * NTOK + n0 + m;
  float s = 0.f, ss = 0.f;
  for (int c = half * 64; c < half * 64 + 64; ++c) {
    float v = xb[(size_t)c * NTOK];
    s += v; ss += v * v;
  }
  s += __shfl_xor(s, 16, 32);
  ss += __shfl_xor(ss, 16, 32);
  float mean = s * (1.f / 128.f);
  float var = ss * (1.f / 128.f) - mean * mean;
  float rstd = rsqrtf(var + 1e-5f);

  __bf16* Am = &At[wave][m * KLDA];
  for (int c = half * 64; c < half * 64 + 64; ++c) {
    float v = xb[(size_t)c * NTOK];
    Am[c] = (__bf16)(((v - mean) * rstd) * ln_g[c] + ln_b[c]);
  }
  __syncthreads();

  v16bf a0 = frag_a_lds(&At[wave][0], KLDA, 0, lane);
  v16bf a1 = frag_a_lds(&At[wave][0], KLDA, 32, lane);
  v16bf a2 = frag_a_lds(&At[wave][0], KLDA, 64, lane);
  v16bf a3 = frag_a_lds(&At[wave][0], KLDA, 96, lane);

  const __bf16* Ws[3] = {wqb, wkb, wvb};
  float* Out[3] = {Qc, Kc, Vc};
#pragma unroll
  for (int w = 0; w < 3; ++w) {
#pragma unroll
    for (int nt = 0; nt < 8; ++nt) {
      v8f acc = {};
      acc = wmma_bf16(a0, frag_b_w(Ws[w], CCH, 0, nt * 16, lane), acc);
      acc = wmma_bf16(a1, frag_b_w(Ws[w], CCH, 32, nt * 16, lane), acc);
      acc = wmma_bf16(a2, frag_b_w(Ws[w], CCH, 64, nt * 16, lane), acc);
      acc = wmma_bf16(a3, frag_b_w(Ws[w], CCH, 96, nt * 16, lane), acc);
      int oc = nt * 16 + m;
      float* op = Out[w] + (size_t)(b * CCH + oc) * NTOK + n0 + half * 8;
      *(float4*)op       = make_float4(acc[0], acc[1], acc[2], acc[3]);
      *(float4*)(op + 4) = make_float4(acc[4], acc[5], acc[6], acc[7]);
    }
  }
}

// ---------------------------------------------------------------------------
// Per-(b,c) token-softmax stats for K: running (max, sum·e^max) online softmax.
// TDM double-buffers 16KB chunks of the channel row into LDS; wave 0 issues
// the DMA for chunk c+1 while all 8 waves reduce chunk c from LDS.
// ---------------------------------------------------------------------------
#define KST_CHUNK 4096  // f32 elements per chunk (16 KB)

__global__ void k_kstats(const float* __restrict__ Kc, float* kmax, float* krcp) {
  __shared__ float buf[2][KST_CHUNK];
  __shared__ float redm[256], reds[256];
  int bc = blockIdx.x;
  int tid = threadIdx.x;
  const float* p = Kc + (size_t)bc * NTOK;
  const int nchunk = NTOK / KST_CHUNK;
  bool wave0 = (tid >> 5) == 0;
  (void)wave0;

  float mx = -1e30f, sum = 0.f;

#if HAVE_TDM
  if (wave0) tdm_load_1d(lds_byte_off(&buf[0][0]), p, KST_CHUNK);
  for (int c = 0; c < nchunk; ++c) {
    if (wave0) {
      if (c + 1 < nchunk) {
        tdm_load_1d(lds_byte_off(&buf[(c + 1) & 1][0]), p + (size_t)(c + 1) * KST_CHUNK,
                    KST_CHUNK);
        __builtin_amdgcn_s_wait_tensorcnt(1);  // chunk c done; c+1 may stay in flight
      } else {
        __builtin_amdgcn_s_wait_tensorcnt(0);  // drain for the final chunk
      }
    }
    __syncthreads();  // chunk c resident in LDS for all waves
    const float* s = &buf[c & 1][0];
    for (int i = tid; i < KST_CHUNK; i += 256) {
      float v = s[i];
      float nm = fmaxf(mx, v);
      sum = sum * __expf(mx - nm) + __expf(v - nm);
      mx = nm;
    }
    __syncthreads();  // everyone done with buf[c&1] before its DMA overwrite
  }
#else
  for (int i = tid; i < NTOK; i += 256) {
    float v = p[i];
    float nm = fmaxf(mx, v);
    sum = sum * __expf(mx - nm) + __expf(v - nm);
    mx = nm;
  }
#endif

  redm[tid] = mx;
  reds[tid] = sum;
  __syncthreads();
  for (int st = 128; st > 0; st >>= 1) {
    if (tid < st) {
      float m2 = redm[tid + st], s2 = reds[tid + st];
      float M = fmaxf(redm[tid], m2);
      reds[tid] = reds[tid] * __expf(redm[tid] - M) + s2 * __expf(m2 - M);
      redm[tid] = M;
    }
    __syncthreads();
  }
  if (tid == 0) { kmax[bc] = redm[0]; krcp[bc] = 1.f / reds[0]; }
}

// ---------------------------------------------------------------------------
// ctx[b,h,dk,dv] = sum_n softmax_k[dk,n] * v[dv,n]  via WMMA (K reduced 32/iter)
// ---------------------------------------------------------------------------
__global__ void k_ctx(const float* __restrict__ Kc, const float* __restrict__ Vc,
                      const float* __restrict__ kmax, const float* __restrict__ krcp,
                      float* __restrict__ ctx) {
  int bh = blockIdx.x;
  int b = bh >> 3, h = bh & 7;
  int lane = threadIdx.x & 31, wave = threadIdx.x >> 5;
  int m = lane & 15, half = lane >> 4;
  int kb = half * 8, kh = half * 16;

  const float* Kp = Kc + (size_t)(b * CCH + h * 16 + m) * NTOK;
  const float* Vp = Vc + (size_t)(b * CCH + h * 16 + m) * NTOK;
  float km = kmax[b * CCH + h * 16 + m];

  v8f acc = {};
  for (int chunk = wave; chunk < NTOK / 32; chunk += 8) {
    int n0 = chunk * 32;
    UF fa, fb;
#pragma unroll
    for (int j = 0; j < 4; ++j) {
      float2 p0 = *(const float2*)(Kp + n0 + kb + 2 * j);
      float2 p1 = *(const float2*)(Kp + n0 + 16 + kb + 2 * j);
      fa.u[j]     = pack2(__expf(p0.x - km), __expf(p0.y - km));
      fa.u[j + 4] = pack2(__expf(p1.x - km), __expf(p1.y - km));
    }
#pragma unroll
    for (int j = 0; j < 8; ++j) {
      float2 p = *(const float2*)(Vp + n0 + kb + 2 * j + (j >= 4 ? 8 : 0));
      fb.u[j] = pack2(p.x, p.y);
    }
    acc = wmma_bf16(fa.v, fb.v, acc);
  }

  __shared__ float red[8][256];
#pragma unroll
  for (int j = 0; j < 8; ++j) red[wave][lane * 8 + j] = acc[j];
  __syncthreads();
  if (wave == 0) {
#pragma unroll
    for (int j = 0; j < 8; ++j) {
      float sv = 0.f;
      for (int w2 = 0; w2 < 8; ++w2) sv += red[w2][lane * 8 + j];
      int row = j + 8 * half, col = m;  // row = dk, col = dv
      ctx[((b * 8 + h) * 16 + row) * 16 + col] = sv * krcp[b * CCH + h * 16 + row];
    }
  }
}

// ---------------------------------------------------------------------------
// Q softmax (over head dims) + ctx matvec + w_rep GEMM + residual.
// Writes attn_skip channel-major.
// ---------------------------------------------------------------------------
__global__ void k_attend(const float* __restrict__ x, const float* __restrict__ Qc,
                         const float* __restrict__ ctx, const float* __restrict__ gamma,
                         const __bf16* __restrict__ wrepb, const float* __restrict__ b_rep,
                         float* __restrict__ Skip) {
  __shared__ __bf16 At[8][16 * KLDA];
  __shared__ float ctxs[8 * 16 * 16];
  int lane = threadIdx.x & 31, wave = threadIdx.x >> 5;
  int tile = blockIdx.x * 8 + wave;
  int g0 = tile * 16;
  int b = g0 >> 15, n0 = g0 & (NTOK - 1);
  int m = lane & 15, half = lane >> 4;

  for (int i = threadIdx.x; i < 2048; i += 256) ctxs[i] = ctx[b * 2048 + i];
  __syncthreads();

  __bf16* Am = &At[wave][m * KLDA];
#pragma unroll
  for (int hh = 0; hh < 4; ++hh) {
    int h = half * 4 + hh;
    const float* qp = Qc + (size_t)(b * CCH + h * 16) * NTOK + n0 + m;
    float q[16];
    float mx = -1e30f;
#pragma unroll
    for (int dk = 0; dk < 16; ++dk) { q[dk] = qp[(size_t)dk * NTOK]; mx = fmaxf(mx, q[dk]); }
    float sum = 0.f;
#pragma unroll
    for (int dk = 0; dk < 16; ++dk) { q[dk] = __expf(q[dk] - mx); sum += q[dk]; }
    float r = 1.f / sum;
    const float* cx = ctxs + h * 256;
#pragma unroll
    for (int dv = 0; dv < 16; ++dv) {
      float av = 0.f;
#pragma unroll
      for (int dk = 0; dk < 16; ++dk) av += q[dk] * cx[dk * 16 + dv];
      Am[h * 16 + dv] = (__bf16)(av * r);
    }
  }
  __syncthreads();

  v16bf a0 = frag_a_lds(&At[wave][0], KLDA, 0, lane);
  v16bf a1 = frag_a_lds(&At[wave][0], KLDA, 32, lane);
  v16bf a2 = frag_a_lds(&At[wave][0], KLDA, 64, lane);
  v16bf a3 = frag_a_lds(&At[wave][0], KLDA, 96, lane);

#pragma unroll
  for (int nt = 0; nt < 8; ++nt) {
    v8f acc = {};
    acc = wmma_bf16(a0, frag_b_w(wrepb, CCH, 0, nt * 16, lane), acc);
    acc = wmma_bf16(a1, frag_b_w(wrepb, CCH, 32, nt * 16, lane), acc);
    acc = wmma_bf16(a2, frag_b_w(wrepb, CCH, 64, nt * 16, lane), acc);
    acc = wmma_bf16(a3, frag_b_w(wrepb, CCH, 96, nt * 16, lane), acc);
    int oc = nt * 16 + m;
    float gg = gamma[oc], br = b_rep[oc];
    const float* xp = x + (size_t)(b * CCH + oc) * NTOK + n0 + half * 8;
    float* sp = Skip + (size_t)(b * CCH + oc) * NTOK + n0 + half * 8;
    float4 x0 = *(const float4*)xp, x1 = *(const float4*)(xp + 4);
    float4 o0, o1;
    o0.x = x0.x + gg * (acc[0] + br);
    o0.y = x0.y + gg * (acc[1] + br);
    o0.z = x0.z + gg * (acc[2] + br);
    o0.w = x0.w + gg * (acc[3] + br);
    o1.x = x1.x + gg * (acc[4] + br);
    o1.y = x1.y + gg * (acc[5] + br);
    o1.z = x1.z + gg * (acc[6] + br);
    o1.w = x1.w + gg * (acc[7] + br);
    *(float4*)sp       = o0;
    *(float4*)(sp + 4) = o1;
  }
}

// ---------------------------------------------------------------------------
// 3x3x3 conv (SAME, C=128->128) as implicit GEMM.  One block = 16 x-positions
// of one (b,z,y) row; 8 waves each own one 16-oc tile; 27 taps x 4 K-chunks
// of WMMA.  Accumulates BN sum/sumsq per channel.  Optionally fuses
// bn-scale/shift + LeakyReLU into the input staging (for conv2).
// ---------------------------------------------------------------------------
__global__ void k_conv3(const float* __restrict__ src, const __bf16* __restrict__ wcb,
                        const float* __restrict__ bnscale, const float* __restrict__ bnshift,
                        int fuse, float* __restrict__ dst,
                        float* __restrict__ bsum, float* __restrict__ bsumsq) {
  __shared__ __bf16 S[162 * KLDA];  // [ (dz*3+dy)*18 + x+1 ][ ic ]
  int blk = blockIdx.x;             // b*2048 + z*64 + y*2 + xh
  int b = blk >> 11;
  int rem = blk & 2047;
  int z = rem >> 6;
  int y = (rem >> 1) & 31;
  int xh = rem & 1;
  int xbase = xh * 16;

  __builtin_prefetch(wcb, 0, 1);  // global_prefetch_b8 on the weight stream

  for (int i = threadIdx.x; i < CCH * 162; i += 256) {
    int sp = i % 162, ic = i / 162;
    int dz = sp / 54;
    int r2 = sp - dz * 54;
    int dy = r2 / 18;
    int xx1 = r2 - dy * 18;
    int zz = z + dz - 1, yy = y + dy - 1, xx = xbase + xx1 - 1;
    float v = 0.f;
    if ((unsigned)zz < 32u && (unsigned)yy < 32u && (unsigned)xx < 32u)
      v = src[(size_t)(b * CCH + ic) * NTOK + (zz * 32 + yy) * 32 + xx];
    if (fuse) v = lrelu(v * bnscale[ic] + bnshift[ic]);
    S[sp * KLDA + ic] = (__bf16)v;
  }
  __syncthreads();

  int lane = threadIdx.x & 31, wave = threadIdx.x >> 5;  // wave = oc tile
  int m = lane & 15, half = lane >> 4;
  v8f acc = {};
  for (int tap = 0; tap < 27; ++tap) {
    int dzt = tap / 9, dyt = (tap / 3) % 3, dxt = tap % 3;
    const __bf16* Ab = S + (size_t)((dzt * 3 + dyt) * 18 + dxt) * KLDA;
    const __bf16* Wt = wcb + (size_t)tap * CCH * CCH;
#pragma unroll
    for (int kc = 0; kc < 4; ++kc) {
      v16bf a = frag_a_lds(Ab, KLDA, kc * 32, lane);
      v16bf bf = frag_b_w(Wt, CCH, kc * 32, wave * 16, lane);
      acc = wmma_bf16(a, bf, acc);
    }
  }

  int oc = wave * 16 + m;
  float* op = dst + (size_t)(b * CCH + oc) * NTOK + (z * 32 + y) * 32 + xbase + half * 8;
  *(float4*)op       = make_float4(acc[0], acc[1], acc[2], acc[3]);
  *(float4*)(op + 4) = make_float4(acc[4], acc[5], acc[6], acc[7]);

  float s = 0.f, q = 0.f;
#pragma unroll
  for (int j = 0; j < 8; ++j) { s += acc[j]; q += acc[j] * acc[j]; }
  s += __shfl_xor(s, 16, 32);
  q += __shfl_xor(q, 16, 32);
  if (half == 0) {
    atomicAdd(&bsum[oc], s);
    atomicAdd(&bsumsq[oc], q);
  }
}

// ---------------------------------------------------------------------------
// BN finalize: per-channel scale/shift from batch sums
// ---------------------------------------------------------------------------
__global__ void k_bnfin(const float* __restrict__ bsum, const float* __restrict__ bsq,
                        const float* __restrict__ g, const float* __restrict__ bta,
                        float* scale, float* shift) {
  int c = threadIdx.x;
  const float cnt = (float)MTOT;
  float mean = bsum[c] / cnt;
  float var = bsq[c] / cnt - mean * mean;
  float sc = g[c] * rsqrtf(var + 1e-5f);
  scale[c] = sc;
  shift[c] = bta[c] - mean * sc;
}

// ---------------------------------------------------------------------------
// Final: out = lrelu(bn2(T2) + skip);  y = skip + out @ w8^T + b8
// ---------------------------------------------------------------------------
__global__ void k_final(const float* __restrict__ T2, const float* __restrict__ Skip,
                        const float* __restrict__ s2, const float* __restrict__ h2,
                        const __bf16* __restrict__ w8b, const float* __restrict__ b8,
                        float* __restrict__ yout) {
  __shared__ __bf16 At[8][16 * KLDA];
  int lane = threadIdx.x & 31, wave = threadIdx.x >> 5;
  int tile = blockIdx.x * 8 + wave;
  int g0 = tile * 16;
  int b = g0 >> 15, n0 = g0 & (NTOK - 1);
  int m = lane & 15, half = lane >> 4;

  __bf16* Am = &At[wave][m * KLDA];
  for (int c = half * 64; c < half * 64 + 64; ++c) {
    size_t idx = (size_t)(b * CCH + c) * NTOK + n0 + m;
    float v = T2[idx] * s2[c] + h2[c] + Skip[idx];
    Am[c] = (__bf16)lrelu(v);
  }
  __syncthreads();

  v16bf a0 = frag_a_lds(&At[wave][0], KLDA, 0, lane);
  v16bf a1 = frag_a_lds(&At[wave][0], KLDA, 32, lane);
  v16bf a2 = frag_a_lds(&At[wave][0], KLDA, 64, lane);
  v16bf a3 = frag_a_lds(&At[wave][0], KLDA, 96, lane);

#pragma unroll
  for (int nt = 0; nt < 8; ++nt) {
    v8f acc = {};
    acc = wmma_bf16(a0, frag_b_w(w8b, CCH, 0, nt * 16, lane), acc);
    acc = wmma_bf16(a1, frag_b_w(w8b, CCH, 32, nt * 16, lane), acc);
    acc = wmma_bf16(a2, frag_b_w(w8b, CCH, 64, nt * 16, lane), acc);
    acc = wmma_bf16(a3, frag_b_w(w8b, CCH, 96, nt * 16, lane), acc);
    int oc = nt * 16 + m;
    float bb = b8[oc];
    const float* sp = Skip + (size_t)(b * CCH + oc) * NTOK + n0 + half * 8;
    float* op = yout + (size_t)(b * CCH + oc) * NTOK + n0 + half * 8;
    float4 k0 = *(const float4*)sp, k1 = *(const float4*)(sp + 4);
    *(float4*)op = make_float4(k0.x + acc[0] + bb, k0.y + acc[1] + bb,
                               k0.z + acc[2] + bb, k0.w + acc[3] + bb);
    *(float4*)(op + 4) = make_float4(k1.x + acc[4] + bb, k1.y + acc[5] + bb,
                                     k1.z + acc[6] + bb, k1.w + acc[7] + bb);
  }
}

// ---------------------------------------------------------------------------
extern "C" void kernel_launch(void* const* d_in, const int* in_sizes, int n_in,
                              void* d_out, int out_size, void* d_ws, size_t ws_size,
                              hipStream_t stream) {
  (void)in_sizes; (void)n_in; (void)out_size; (void)ws_size;
  const float* x       = (const float*)d_in[0];
  const float* ln_g    = (const float*)d_in[1];
  const float* ln_b    = (const float*)d_in[2];
  const float* gamma   = (const float*)d_in[3];
  const float* wq      = (const float*)d_in[4];
  const float* wk      = (const float*)d_in[5];
  const float* wv      = (const float*)d_in[6];
  const float* w_rep   = (const float*)d_in[7];
  const float* b_rep   = (const float*)d_in[8];
  const float* conv1_w = (const float*)d_in[9];
  const float* bn1_g   = (const float*)d_in[10];
  const float* bn1_b   = (const float*)d_in[11];
  const float* conv2_w = (const float*)d_in[12];
  const float* bn2_g   = (const float*)d_in[13];
  const float* bn2_b   = (const float*)d_in[14];
  const float* conv8_w = (const float*)d_in[15];
  const float* conv8_b = (const float*)d_in[16];
  float* yout = (float*)d_out;

  char* wsp = (char*)d_ws;
  auto carve = [&](size_t bytes) -> void* {
    void* p = (void*)wsp;
    wsp += (bytes + 255) & ~(size_t)255;
    return p;
  };
  const size_t planar = (size_t)MTOT * CCH * sizeof(float);  // 33.5 MB
  float* Qc   = (float*)carve(planar);
  float* Kc   = (float*)carve(planar);
  float* Vc   = (float*)carve(planar);
  float* Skip = (float*)carve(planar);
  float* T1   = (float*)carve(planar);
  float* T2   = (float*)carve(planar);
  __bf16* wqb   = (__bf16*)carve(CCH * CCH * 2);
  __bf16* wkb   = (__bf16*)carve(CCH * CCH * 2);
  __bf16* wvb   = (__bf16*)carve(CCH * CCH * 2);
  __bf16* wrepb = (__bf16*)carve(CCH * CCH * 2);
  __bf16* w8b   = (__bf16*)carve(CCH * CCH * 2);
  __bf16* c1b   = (__bf16*)carve(27 * CCH * CCH * 2);
  __bf16* c2b   = (__bf16*)carve(27 * CCH * CCH * 2);
  float* kmax = (float*)carve(BATCH * CCH * 4);
  float* krcp = (float*)carve(BATCH * CCH * 4);
  float* ctx  = (float*)carve(BATCH * 8 * 16 * 16 * 4);
  float* bn   = (float*)carve(8 * CCH * 4);  // sum1,sq1,sum2,sq2,s1,h1,s2,h2
  float* bsum1 = bn + 0 * CCH, *bsq1 = bn + 1 * CCH;
  float* bsum2 = bn + 2 * CCH, *bsq2 = bn + 3 * CCH;
  float* s1 = bn + 4 * CCH, *h1 = bn + 5 * CCH;
  float* s2 = bn + 6 * CCH, *h2 = bn + 7 * CCH;

  (void)hipMemsetAsync(bn, 0, 4 * CCH * sizeof(float), stream);

  k_prep<<<dim3((27 * CCH * CCH + 255) / 256), dim3(256), 0, stream>>>(
      wq, wk, wv, w_rep, conv8_w, conv1_w, conv2_w,
      wqb, wkb, wvb, wrepb, w8b, c1b, c2b);

  k_ln_qkv<<<dim3(MTOT / 128), dim3(256), 0, stream>>>(x, ln_g, ln_b, wqb, wkb, wvb,
                                                       Qc, Kc, Vc);
  k_kstats<<<dim3(BATCH * CCH), dim3(256), 0, stream>>>(Kc, kmax, krcp);
  k_ctx<<<dim3(BATCH * 8), dim3(256), 0, stream>>>(Kc, Vc, kmax, krcp, ctx);
  k_attend<<<dim3(MTOT / 128), dim3(256), 0, stream>>>(x, Qc, ctx, gamma, wrepb, b_rep,
                                                       Skip);
  k_conv3<<<dim3(BATCH * 32 * 32 * 2), dim3(256), 0, stream>>>(Skip, c1b, nullptr, nullptr,
                                                               0, T1, bsum1, bsq1);
  k_bnfin<<<dim3(1), dim3(CCH), 0, stream>>>(bsum1, bsq1, bn1_g, bn1_b, s1, h1);
  k_conv3<<<dim3(BATCH * 32 * 32 * 2), dim3(256), 0, stream>>>(T1, c2b, s1, h1,
                                                               1, T2, bsum2, bsq2);
  k_bnfin<<<dim3(1), dim3(CCH), 0, stream>>>(bsum2, bsq2, bn2_g, bn2_b, s2, h2);
  k_final<<<dim3(MTOT / 128), dim3(256), 0, stream>>>(T2, Skip, s2, h2, w8b, conv8_b,
                                                      yout);
}
